// BaseAttentivePool_49263274885766
// MI455X (gfx1250) — compile-verified, matrix operand order checked by reading, not run on hardware
//
#include <hip/hip_runtime.h>
#include <hip/hip_bf16.h>
#include <math.h>

// ---------------------------------------------------------------------------
// MI455X (gfx1250) wave32 implementation of segment-softmax attention pooling.
// GEMMs: v_wmma_f32_16x16x32_f16 (f16 operands, f32 accum) -> HBM-bound.
// Weight tiles staged to LDS with the Tensor Data Mover (tensor_load_to_lds),
// synchronized via s_wait_tensorcnt.
// ---------------------------------------------------------------------------

typedef _Float16     v16h __attribute__((ext_vector_type(16)));
typedef float        v8f  __attribute__((ext_vector_type(8)));
typedef unsigned int v4u  __attribute__((ext_vector_type(4)));
typedef unsigned int v8u  __attribute__((ext_vector_type(8)));

#define WAVES 8   // waves per 256-thread block; each wave owns a 16-row tile

__device__ inline v8f wmma_f16(v16h a, v16h b, v8f c) {
  // 8 args: (neg_a, A, neg_b, B, c_mod, C, reuse_a, reuse_b)
  return __builtin_amdgcn_wmma_f32_16x16x32_f16(false, a, false, b,
                                                (short)0, c, false, false);
}

// ---------------------------------------------------------------------------
// Tensor Data Mover: DMA a strided 2D f32 tile (tile_w x tile_h rows) from a
// row-major global tensor (row stride stride_w elements) into LDS at lds_off.
// D# layout per CDNA5 ISA.  Issued once per wave
// (EXEC ignored); caller waits TENSORcnt then barriers.
// ---------------------------------------------------------------------------
#if defined(__gfx1250__)
__device__ inline void tdm_load_2d_f32(unsigned lds_off, const void* gptr,
                                       unsigned tensor_w, unsigned tensor_h,
                                       unsigned tile_w, unsigned tile_h,
                                       unsigned stride_w) {
  unsigned long long ga = (unsigned long long)(uintptr_t)gptr;
  v4u g0;
  g0[0] = 1u;                                    // count=1, user descriptor
  g0[1] = lds_off;                               // lds_addr (bytes)
  g0[2] = (unsigned)ga;                          // global_addr[31:0]
  g0[3] = ((unsigned)(ga >> 32) & 0x01FFFFFFu)   // global_addr[56:32]
          | 0x80000000u;                         // type=2 ("image")
  v8u g1;
  g1[0] = 0x00020000u;                           // data_size=2 -> 4 bytes
  g1[1] = (tensor_w & 0xFFFFu) << 16;            // tensor_dim0[15:0]
  g1[2] = (tensor_w >> 16) | ((tensor_h & 0xFFFFu) << 16);  // dim0[31:16]|dim1
  g1[3] = (tensor_h >> 16) | ((tile_w & 0xFFFFu) << 16);    // dim1[31:16]|tile0
  g1[4] = tile_h & 0xFFFFu;                      // tile_dim1 (tile_dim2 = 0)
  g1[5] = stride_w;                              // tensor_dim0_stride[31:0]
  g1[6] = 0u;                                    // stride hi / dim1_stride lo
  g1[7] = 0u;
  asm volatile("tensor_load_to_lds %0, %1" :: "s"(g0), "s"(g1) : "memory");
}
#endif

// Stage a 2D f32 weight tile into LDS: TDM on gfx1250 (wave 0 issues, waits
// TENSORcnt), plain cooperative loads otherwise.  Must be followed by
// __syncthreads() at the call site.
__device__ inline void stage_weights(float* lds_dst, const float* g_src,
                                     int tensor_w, int tile_w, int tile_h,
                                     int tid) {
#if defined(__gfx1250__)
  if (tid < 32) {
    tdm_load_2d_f32((unsigned)(uintptr_t)lds_dst, g_src, (unsigned)tensor_w,
                    (unsigned)tile_h, (unsigned)tile_w, (unsigned)tile_h,
                    (unsigned)tensor_w);
    __builtin_amdgcn_s_wait_tensorcnt(0);
  }
#else
  for (int i = tid; i < tile_w * tile_h; i += 256)
    lds_dst[i] = g_src[(i / tile_w) * tensor_w + (i % tile_w)];
#endif
}

// A fragment (16x32 f16, MxK) from row-major f32 matrix.
// Lane L (m = row0 + (L&15)) holds K chunks [klo,klo+8) and [klo+16,klo+24),
// klo = kb + (L>=16 ? 8 : 0)   (ISA 7.12.2, 16-bit A 16x32 table).
__device__ inline v16h load_a_frag(const float* X, int ld, int row0, int kb,
                                   int maxrow) {
  int lane = threadIdx.x & 31;
  int m = row0 + (lane & 15);
  if (m > maxrow) m = maxrow;
  const float* rp = X + (size_t)m * ld + kb + ((lane & 16) ? 8 : 0);
  v16h a;
#pragma unroll
  for (int j = 0; j < 8; ++j) a[j] = (_Float16)rp[j];
#pragma unroll
  for (int j = 0; j < 8; ++j) a[j + 8] = (_Float16)rp[16 + j];
  return a;
}

// B fragment (32x16 f16, KxN) from row-major f32 weights in LDS (stride NW),
// converting f32->f16 at fragment-build time.
__device__ inline v16h load_b_frag(const float* W, int NW, int nb, int kb) {
  int lane = threadIdx.x & 31;
  int n = nb + (lane & 15);
  int klo = kb + ((lane & 16) ? 8 : 0);
  v16h b;
#pragma unroll
  for (int j = 0; j < 8; ++j) b[j] = (_Float16)W[(klo + j) * NW + n];
#pragma unroll
  for (int j = 0; j < 8; ++j) b[j + 8] = (_Float16)W[(klo + 16 + j) * NW + n];
  return b;
}

// Float atomic max via sign-bucketed integer atomics (works with -inf init).
__device__ inline void atomicMaxFloat(float* addr, float val) {
  if (val >= 0.0f)
    atomicMax((int*)addr, __float_as_int(val));
  else
    atomicMin((unsigned int*)addr, __float_as_uint(val));
}

// ---------------------------------------------------------------------------
// Kernel 1: re-initialize accumulators every launch (graph replays!).
// ---------------------------------------------------------------------------
__global__ void init_kernel(float* out, float* mArr, float* sArr, int NP) {
  int i = blockIdx.x * blockDim.x + threadIdx.x;
  if (i < NP * 64) out[i] = 0.0f;
  if (i < NP * 4) { sArr[i] = 0.0f; mArr[i] = -INFINITY; }
}

// ---------------------------------------------------------------------------
// Kernel 2: q_scaled = (x_parent @ wq + bq) * DQK^-0.5     (NP x 32)
// ---------------------------------------------------------------------------
__global__ __launch_bounds__(256) void qproj_kernel(
    const float* __restrict__ xp, const float* __restrict__ wq,
    const float* __restrict__ bq, float* __restrict__ qs, int NP) {
  __shared__ float sW[64 * 32];
  __shared__ float sB[32];
  int tid = threadIdx.x;
  stage_weights(sW, wq, 32, 32, 64, tid);          // dense 64x32 tensor
  if (tid < 32) sB[tid] = bq[tid];
  __syncthreads();

  int w = tid >> 5, lane = tid & 31;
  int r0 = (blockIdx.x * WAVES + w) * 16;
  if (r0 >= NP) return;
  int mx = NP - 1;
  bool full = (r0 + 16 <= NP);
  const float SCALE = 0.35355339059327373f;  // 8^-0.5

  v16h a0 = load_a_frag(xp, 64, r0, 0, mx);
  v16h a1 = load_a_frag(xp, 64, r0, 32, mx);
#pragma unroll
  for (int t = 0; t < 2; ++t) {
    v8f acc = {};
    acc = wmma_f16(a0, load_b_frag(sW, 32, t * 16, 0), acc);
    acc = wmma_f16(a1, load_b_frag(sW, 32, t * 16, 32), acc);
    int n = t * 16 + (lane & 15);
    float bn = sB[n];
    int mbase = r0 + ((lane & 16) ? 8 : 0);
    if (full) {
#pragma unroll
      for (int r = 0; r < 8; ++r)
        qs[(size_t)(mbase + r) * 32 + n] = (acc[r] + bn) * SCALE;
    } else {
#pragma unroll
      for (int r = 0; r < 8; ++r)
        if (mbase + r < NP)
          qs[(size_t)(mbase + r) * 32 + n] = (acc[r] + bn) * SCALE;
    }
  }
}

// ---------------------------------------------------------------------------
// Kernel 3: per child tile -> k = x@Wk + bk + rpe_k ; q = qs[parent] + rpe_q ;
// compat[c][h] = <q,k>_head ; segment max via float atomic max.
// ---------------------------------------------------------------------------
__global__ __launch_bounds__(256) void compat_kernel(
    const float* __restrict__ xc, const float* __restrict__ ea,
    const int* __restrict__ idx, const float* __restrict__ wkv,
    const float* __restrict__ bkv, const float* __restrict__ wk_rpe,
    const float* __restrict__ bk_rpe, const float* __restrict__ wq_rpe,
    const float* __restrict__ bq_rpe, const float* __restrict__ qs,
    float* __restrict__ mArr, float* __restrict__ compat, int NC) {
  __shared__ float sWk[64 * 32];               // Wkv[:, :32] tile via TDM
  __shared__ float sBk[32], sBkr[32], sBqr[32];
  __shared__ float sWkr[9 * 32], sWqr[9 * 32];
  __shared__ float kbuf[WAVES][16 * 32];
  __shared__ float qbuf[WAVES][16 * 32];
  __shared__ float sEA[WAVES][16 * 10];        // padded 9->10
  __shared__ int   sP[WAVES][16];

  int tid = threadIdx.x;
  stage_weights(sWk, wkv, 96, 32, 64, tid);    // 64x32 strided tile of 64x96
  for (int i = tid; i < 9 * 32; i += 256) { sWkr[i] = wk_rpe[i]; sWqr[i] = wq_rpe[i]; }
  if (tid < 32) { sBk[tid] = bkv[tid]; sBkr[tid] = bk_rpe[tid]; sBqr[tid] = bq_rpe[tid]; }
  __syncthreads();

  int w = tid >> 5, lane = tid & 31;
  int c0 = (blockIdx.x * WAVES + w) * 16;
  if (c0 >= NC) return;
  int mx = NC - 1;

  // prefetch next tile of the streaming x_child read (global_prefetch_b8)
  __builtin_prefetch(xc + (size_t)min(c0 + 16 * WAVES, mx) * 64, 0, 1);

  if (lane < 16) sP[w][lane] = idx[min(c0 + lane, mx)];
  for (int i = lane; i < 144; i += 32) {
    int m = i / 9, j = i - m * 9;
    sEA[w][m * 10 + j] = ea[(size_t)min(c0 + m, mx) * 9 + j];
  }

  // k linear part via WMMA -> kbuf (+ bias)
  v16h a0 = load_a_frag(xc, 64, c0, 0, mx);
  v16h a1 = load_a_frag(xc, 64, c0, 32, mx);
#pragma unroll
  for (int t = 0; t < 2; ++t) {
    v8f acc = {};
    acc = wmma_f16(a0, load_b_frag(sWk, 32, t * 16, 0), acc);
    acc = wmma_f16(a1, load_b_frag(sWk, 32, t * 16, 32), acc);
    int n = t * 16 + (lane & 15);
    float bn = sBk[n];
    int mbase = (lane & 16) ? 8 : 0;
#pragma unroll
    for (int r = 0; r < 8; ++r) kbuf[w][(mbase + r) * 32 + n] = acc[r] + bn;
  }
  __builtin_amdgcn_wave_barrier();  // per-wave LDS ops are in order (DScnt)

  // RPE add (K=9 -> VALU, too small for WMMA) + gathered-q build.
  // entry = 32*i + lane -> m=i, n=lane: qs gather is a coalesced row read.
#pragma unroll
  for (int i = 0; i < 16; ++i) {
    int m = i, n = lane;
    float rk = 0.0f, rq = 0.0f;
#pragma unroll
    for (int j = 0; j < 9; ++j) {
      float e = sEA[w][m * 10 + j];
      rk = fmaf(e, sWkr[j * 32 + n], rk);
      rq = fmaf(e, sWqr[j * 32 + n], rq);
    }
    kbuf[w][m * 32 + n] += rk + sBkr[n];
    qbuf[w][m * 32 + n] = qs[(size_t)sP[w][m] * 32 + n] + rq + sBqr[n];
  }
  __builtin_amdgcn_wave_barrier();

  // compat per (child m, head h): 64 pairs / tile, 2 per lane
#pragma unroll
  for (int u = 0; u < 2; ++u) {
    int e = lane + 32 * u;
    int m = e >> 2, h = e & 3;
    float c = 0.0f;
#pragma unroll
    for (int d = 0; d < 8; ++d)
      c = fmaf(qbuf[w][m * 32 + h * 8 + d], kbuf[w][m * 32 + h * 8 + d], c);
    int ch = c0 + m;
    if (ch < NC) {
      compat[(size_t)ch * 4 + h] = c;
      atomicMaxFloat(&mArr[(size_t)sP[w][m] * 4 + h], c);
    }
  }
}

// ---------------------------------------------------------------------------
// Kernel 4: e = exp(compat - segmax[parent]); segment sum via atomicAdd.
// ---------------------------------------------------------------------------
__global__ void expsum_kernel(const float* __restrict__ compat,
                              const int* __restrict__ idx,
                              const float* __restrict__ mArr,
                              float* __restrict__ sArr, int NC) {
  int i = blockIdx.x * blockDim.x + threadIdx.x;
  if (i >= NC * 4) return;
  int c = i >> 2, h = i & 3;
  int p = idx[c];
  float e = __expf(compat[i] - mArr[(size_t)p * 4 + h]);
  atomicAdd(&sArr[(size_t)p * 4 + h], e);
}

// ---------------------------------------------------------------------------
// Kernel 5: v = x_child @ Wv + bv (WMMA, Wv = wkv[:,32:96]);
// out[parent] += v * attn   (head h == 16-wide N-tile t).
// ---------------------------------------------------------------------------
__global__ __launch_bounds__(256) void out_kernel(
    const float* __restrict__ xc, const int* __restrict__ idx,
    const float* __restrict__ wkv, const float* __restrict__ bkv,
    const float* __restrict__ compat, const float* __restrict__ mArr,
    const float* __restrict__ sArr, float* __restrict__ out, int NC) {
  __shared__ float sWv[64 * 64];               // Wkv[:, 32:96] tile via TDM
  __shared__ float sBv[64];
  __shared__ float sAttn[WAVES][16 * 4];
  __shared__ int   sP[WAVES][16];

  int tid = threadIdx.x;
  stage_weights(sWv, wkv + 32, 96, 64, 64, tid);
  if (tid < 64) sBv[tid] = bkv[32 + tid];
  __syncthreads();

  int w = tid >> 5, lane = tid & 31;
  int c0 = (blockIdx.x * WAVES + w) * 16;
  if (c0 >= NC) return;
  int mx = NC - 1;
  bool full = (c0 + 16 <= NC);

  __builtin_prefetch(xc + (size_t)min(c0 + 16 * WAVES, mx) * 64, 0, 1);

  if (lane < 16) sP[w][lane] = idx[min(c0 + lane, mx)];
  __builtin_amdgcn_wave_barrier();

#pragma unroll
  for (int u = 0; u < 2; ++u) {
    int e = lane + 32 * u;
    int m = e >> 2, h = e & 3;
    int c = min(c0 + m, mx);
    int p = sP[w][m];
    sAttn[w][m * 4 + h] =
        __expf(compat[(size_t)c * 4 + h] - mArr[(size_t)p * 4 + h]) /
        (sArr[(size_t)p * 4 + h] + 1e-16f);
  }
  __builtin_amdgcn_wave_barrier();

  v16h a0 = load_a_frag(xc, 64, c0, 0, mx);
  v16h a1 = load_a_frag(xc, 64, c0, 32, mx);
  int nl = lane & 15;
  int mbase = (lane & 16) ? 8 : 0;
#pragma unroll
  for (int t = 0; t < 4; ++t) {   // N-tile t == head t (v is (Nc,4,16))
    v8f acc = {};
    acc = wmma_f16(a0, load_b_frag(sWv, 64, t * 16, 0), acc);
    acc = wmma_f16(a1, load_b_frag(sWv, 64, t * 16, 32), acc);
    int n = t * 16 + nl;
    float bn = sBv[n];
    if (full) {
#pragma unroll
      for (int r = 0; r < 8; ++r) {
        int m = mbase + r;
        atomicAdd(&out[(size_t)sP[w][m] * 64 + n],
                  (acc[r] + bn) * sAttn[w][m * 4 + t]);
      }
    } else {
#pragma unroll
      for (int r = 0; r < 8; ++r) {
        int m = mbase + r;
        if (c0 + m < NC)
          atomicAdd(&out[(size_t)sP[w][m] * 64 + n],
                    (acc[r] + bn) * sAttn[w][m * 4 + t]);
      }
    }
  }
}

// ---------------------------------------------------------------------------
extern "C" void kernel_launch(void* const* d_in, const int* in_sizes, int n_in,
                              void* d_out, int out_size, void* d_ws,
                              size_t ws_size, hipStream_t stream) {
  const float* x_child  = (const float*)d_in[0];
  const float* x_parent = (const float*)d_in[1];
  const int*   index    = (const int*)d_in[2];
  const float* edge_attr= (const float*)d_in[3];
  const float* wq       = (const float*)d_in[4];
  const float* bq       = (const float*)d_in[5];
  const float* wkv      = (const float*)d_in[6];
  const float* bkv      = (const float*)d_in[7];
  const float* wk_rpe   = (const float*)d_in[8];
  const float* bk_rpe   = (const float*)d_in[9];
  const float* wq_rpe   = (const float*)d_in[10];
  const float* bq_rpe   = (const float*)d_in[11];
  float* out = (float*)d_out;

  const int NC = in_sizes[0] / 64;
  const int NP = in_sizes[1] / 64;

  // workspace: qs (NP*32) | segmax (NP*4) | segsum (NP*4) | compat (NC*4)
  float* qs     = (float*)d_ws;
  float* mArr   = qs + (size_t)NP * 32;
  float* sArr   = mArr + (size_t)NP * 4;
  float* compat = sArr + (size_t)NP * 4;

  init_kernel<<<(NP * 64 + 255) / 256, 256, 0, stream>>>(out, mArr, sArr, NP);

  int ptiles = (NP + 15) / 16;
  qproj_kernel<<<(ptiles + WAVES - 1) / WAVES, 256, 0, stream>>>(
      x_parent, wq, bq, qs, NP);

  int ctiles = (NC + 15) / 16;
  compat_kernel<<<(ctiles + WAVES - 1) / WAVES, 256, 0, stream>>>(
      x_child, edge_attr, index, wkv, bkv, wk_rpe, bk_rpe, wq_rpe, bq_rpe, qs,
      mArr, compat, NC);

  expsum_kernel<<<(NC * 4 + 255) / 256, 256, 0, stream>>>(compat, index, mArr,
                                                          sArr, NC);

  out_kernel<<<(ctiles + WAVES - 1) / WAVES, 256, 0, stream>>>(
      x_child, index, wkv, bkv, compat, mArr, sArr, out, NC);
}
